// Sage_Classifier_29755533426830
// MI455X (gfx1250) — compile-verified, matrix operand order checked by reading, not run on hardware
//
#include <hip/hip_runtime.h>

typedef __attribute__((ext_vector_type(16))) __bf16 v16bf;
typedef __attribute__((ext_vector_type(8)))  __bf16 v8bf;
typedef __attribute__((ext_vector_type(8)))  float  v8f;
typedef __attribute__((ext_vector_type(4)))  float  v4f;

#define NV    16384
#define NHID  128
#define NEMB  256
#define NCLS  40
#define NCLSP 48      // class dim padded to 3x16 WMMA tiles

// ---------- fragment packing helpers ----------
// A/B 16-bit fragment: 16 bf16 per lane. Lanes 0-15 hold K={0..7,16..23},
// lanes 16-31 hold K={8..15,24..31} (handled by per-lane 'off' = 0 or 8).
static __device__ __forceinline__ v16bf pack4f(v4f a0, v4f a1, v4f a2, v4f a3) {
  v16bf r;
#pragma unroll
  for (int i = 0; i < 4; ++i) {
    r[i]      = (__bf16)a0[i];
    r[4 + i]  = (__bf16)a1[i];
    r[8 + i]  = (__bf16)a2[i];
    r[12 + i] = (__bf16)a3[i];
  }
  return r;
}
static __device__ __forceinline__ v16bf pack2b(v8bf b0, v8bf b1) {
  v16bf r;
#pragma unroll
  for (int i = 0; i < 8; ++i) { r[i] = b0[i]; r[8 + i] = b1[i]; }
  return r;
}

// ---------- kernel 1: bf16 conversions / transposes ----------
__global__ void prep_kernel(const float* __restrict__ x,
                            const float* __restrict__ Wp,
                            const float* __restrict__ Wm,
                            __bf16* __restrict__ x_bf,
                            __bf16* __restrict__ xT,
                            __bf16* __restrict__ Wp_bf,
                            __bf16* __restrict__ Wm_bf) {
  const int T1 = NV * NHID;          // x elements
  const int T2 = NEMB * (2 * NHID);  // W_proj elements
  const int T3 = NCLSP * NEMB;       // padded W_mlp elements
  int i = blockIdx.x * blockDim.x + threadIdx.x;
  if (i < T1) {
    float v = x[i];
    x_bf[i] = (__bf16)v;                      // row-major bf16 x
    int k = i / NHID, n = i % NHID;
    xT[(size_t)n * NV + k] = (__bf16)v;       // transposed bf16 x (B-side)
  } else if (i < T1 + T2) {
    int j = i - T1;
    Wp_bf[j] = (__bf16)Wp[j];
  } else if (i < T1 + T2 + T3) {
    int j = i - T1 - T2;
    int row = j / NEMB;
    Wm_bf[j] = (row < NCLS) ? (__bf16)Wm[j] : (__bf16)0.0f;
  }
}

// ---------- kernel 2: neigh = (adj @ x) / (rowsum(adj)+1), fused deg ----------
__global__ void __launch_bounds__(256) sage_agg_kernel(
    const float* __restrict__ adj,
    const __bf16* __restrict__ xT,
    __bf16* __restrict__ neigh) {
  __shared__ float sdeg[8 * 16];
  const int lane = threadIdx.x & 31;
  const int wid  = threadIdx.x >> 5;
  const int lm   = lane & 15;
  const int off  = (lane >> 4) << 3;          // 0 or 8 per half-wave
  const int rb   = (blockIdx.x * 8 + wid) * 16;

  const float*  arow = adj + (size_t)(rb + lm) * NV;  // this lane's adj row
  const __bf16* xrow = xT + (size_t)lm * NV;          // this lane's B column

  v8f zero = {0.f, 0.f, 0.f, 0.f, 0.f, 0.f, 0.f, 0.f};
  v8f acc[8];
#pragma unroll
  for (int nt = 0; nt < 8; ++nt) acc[nt] = zero;
  float rs = 0.0f;

  for (int kb = 0; kb < NV; kb += 32) {
    // one 128-B cacheline of adj per row per step
    v4f a0 = *(const v4f*)(arow + kb + off);
    v4f a1 = *(const v4f*)(arow + kb + off + 4);
    v4f a2 = *(const v4f*)(arow + kb + 16 + off);
    v4f a3 = *(const v4f*)(arow + kb + 16 + off + 4);
    __builtin_prefetch(arow + kb + 256, 0, 1);        // global_prefetch_b8
#pragma unroll
    for (int i = 0; i < 4; ++i) rs += a0[i] + a1[i] + a2[i] + a3[i];
    v16bf afrag = pack4f(a0, a1, a2, a3);
#pragma unroll
    for (int nt = 0; nt < 8; ++nt) {
      const __bf16* bp = xrow + (size_t)nt * 16 * NV + kb;
      v16bf bfrag = pack2b(*(const v8bf*)(bp + off),
                           *(const v8bf*)(bp + 16 + off));
      acc[nt] = __builtin_amdgcn_wmma_f32_16x16x32_bf16(
          false, afrag, false, bfrag, (short)0, acc[nt], false, false);
    }
  }

  // full row sum = this half-wave partial + mirrored half-wave partial
  float tot = rs + __shfl_xor(rs, 16, 32);
  if (lane < 16) sdeg[wid * 16 + lm] = tot + 1.0f;
  __syncthreads();

#pragma unroll
  for (int nt = 0; nt < 8; ++nt) {
#pragma unroll
    for (int r = 0; r < 8; ++r) {
      int rr = (lane < 16) ? r : (r + 8);              // C-layout row
      float v = acc[nt][r] / sdeg[wid * 16 + rr];
      neigh[(size_t)(rb + rr) * NHID + nt * 16 + lm] = (__bf16)v;
    }
  }
}

// ---------- kernel 3: out = relu([x|neigh] @ Wp^T) @ Wm^T + b ----------
__global__ void __launch_bounds__(256) proj_mlp_kernel(
    const __bf16* __restrict__ x_bf,
    const __bf16* __restrict__ neigh,
    const __bf16* __restrict__ Wp_bf,
    const __bf16* __restrict__ Wm_bf,
    const float* __restrict__ bias,
    float* __restrict__ out) {
  __shared__ __align__(16) __bf16 sh[8][16][264];  // pitch 264 -> conflict-free TR reads
  const int lane = threadIdx.x & 31;
  const int wid  = threadIdx.x >> 5;
  const int lm   = lane & 15;
  const int off  = (lane >> 4) << 3;
  const int rb   = (blockIdx.x * 8 + wid) * 16;

  const __bf16* xr = x_bf  + (size_t)(rb + lm) * NHID;
  const __bf16* nr = neigh + (size_t)(rb + lm) * NHID;

  v8f zero = {0.f, 0.f, 0.f, 0.f, 0.f, 0.f, 0.f, 0.f};
  v8f acc[16];
#pragma unroll
  for (int nt = 0; nt < 16; ++nt) acc[nt] = zero;

  // GEMM2: h[16x256] = [x|neigh] @ Wp^T, K = 256 (tiles 0-3: x, 4-7: neigh)
#pragma unroll
  for (int kt = 0; kt < 8; ++kt) {
    const __bf16* src = (kt < 4) ? (xr + kt * 32) : (nr + (kt - 4) * 32);
    v16bf afrag = pack2b(*(const v8bf*)(src + off),
                         *(const v8bf*)(src + 16 + off));
#pragma unroll
    for (int nt = 0; nt < 16; ++nt) {
      const __bf16* wrow = Wp_bf + (size_t)(nt * 16 + lm) * (2 * NHID) + kt * 32;
      v16bf bfrag = pack2b(*(const v8bf*)(wrow + off),
                           *(const v8bf*)(wrow + 16 + off));
      acc[nt] = __builtin_amdgcn_wmma_f32_16x16x32_bf16(
          false, afrag, false, bfrag, (short)0, acc[nt], false, false);
    }
  }

  // ReLU + C-layout -> A-layout transpose through LDS (bf16)
#pragma unroll
  for (int nt = 0; nt < 16; ++nt) {
#pragma unroll
    for (int r = 0; r < 8; ++r) {
      int rr = (lane < 16) ? r : (r + 8);
      float v = acc[nt][r];
      sh[wid][rr][nt * 16 + lm] = (__bf16)(v > 0.0f ? v : 0.0f);
    }
  }
  __syncthreads();

  // GEMM3: out[16x48] = h2 @ Wm^T, K = 256
  v8f acc3[3];
#pragma unroll
  for (int nt = 0; nt < 3; ++nt) acc3[nt] = zero;
#pragma unroll
  for (int kt = 0; kt < 8; ++kt) {
    const __bf16* lrow = &sh[wid][lm][kt * 32];
    v16bf afrag = pack2b(*(const v8bf*)(lrow + off),
                         *(const v8bf*)(lrow + 16 + off));
#pragma unroll
    for (int nt = 0; nt < 3; ++nt) {
      const __bf16* wrow = Wm_bf + (size_t)(nt * 16 + lm) * NEMB + kt * 32;
      v16bf bfrag = pack2b(*(const v8bf*)(wrow + off),
                           *(const v8bf*)(wrow + 16 + off));
      acc3[nt] = __builtin_amdgcn_wmma_f32_16x16x32_bf16(
          false, afrag, false, bfrag, (short)0, acc3[nt], false, false);
    }
  }

#pragma unroll
  for (int nt = 0; nt < 3; ++nt) {
    int n = nt * 16 + lm;
    if (n < NCLS) {
      float b = bias[n];
#pragma unroll
      for (int r = 0; r < 8; ++r) {
        int rr = (lane < 16) ? r : (r + 8);
        out[(size_t)(rb + rr) * NCLS + n] = acc3[nt][r] + b;
      }
    }
  }
}

extern "C" void kernel_launch(void* const* d_in, const int* in_sizes, int n_in,
                              void* d_out, int out_size, void* d_ws, size_t ws_size,
                              hipStream_t stream) {
  const float* x   = (const float*)d_in[0];
  const float* adj = (const float*)d_in[1];
  const float* Wp  = (const float*)d_in[2];
  const float* Wm  = (const float*)d_in[3];
  const float* bm  = (const float*)d_in[4];
  float* out = (float*)d_out;

  char* ws = (char*)d_ws;
  const size_t S1 = (size_t)NV * NHID * sizeof(__bf16);   // 4 MiB each
  __bf16* x_bf  = (__bf16*)(ws);
  __bf16* xT    = (__bf16*)(ws + S1);
  __bf16* neigh = (__bf16*)(ws + 2 * S1);
  __bf16* Wp_bf = (__bf16*)(ws + 3 * S1);
  __bf16* Wm_bf = (__bf16*)(ws + 3 * S1 + (size_t)NEMB * 2 * NHID * sizeof(__bf16));

  const int prep_total = NV * NHID + NEMB * (2 * NHID) + NCLSP * NEMB;
  prep_kernel<<<(prep_total + 255) / 256, 256, 0, stream>>>(
      x, Wp, Wm, x_bf, xT, Wp_bf, Wm_bf);

  // 1024 row-blocks of 16 rows, 8 waves per workgroup
  sage_agg_kernel<<<NV / (16 * 8), 256, 0, stream>>>(adj, xT, neigh);

  proj_mlp_kernel<<<NV / (16 * 8), 256, 0, stream>>>(
      x_bf, neigh, Wp_bf, Wm_bf, bm, out);
}
